// UMT_65687229825283
// MI455X (gfx1250) — matrix operand order repaired and checked by reading, not compiled, not run on hardware
//
#include <hip/hip_runtime.h>

// ---------------------------------------------------------------------------
// partial-OT loss for MI455X (gfx1250, wave32, WMMA + async global->LDS)
//
//   B=64, m=n=512, d=256.  Per batch:
//     S0 = sum exp(-C/beta), CT = sum C*exp(-C/beta), rs0 = row-0 sum,
//     cs0 = col-0 sum, with C = 0.5(|v_i|^2+|a_j|^2) - v_i.a_j, beta=1000,
//   then a 10-step scalar recurrence and a batch mean.
//
//   v3: workgroup (8 waves) computes a 64x128 output block.  A/B f16 panels
//   are staged into LDS with GLOBAL_LOAD_ASYNC_TO_LDS_B128 (ASYNCcnt) in two
//   K=128 chunks (51KB LDS, 272B padded rows -> 4-bank row skew, conflict-free
//   ds_load_b128 fragment reads).  Each wave computes a 32x32 tile: 4
//   independent f32 accumulators, 4 v_wmma_f32_16x16x32_f16 per k-step.
//   Epilogue (exp + 4 reductions) fused on accumulator registers; one f32
//   atomic per wave per statistic.
// ---------------------------------------------------------------------------

typedef __attribute__((ext_vector_type(16))) _Float16 v16h;
typedef __attribute__((ext_vector_type(8)))  _Float16 v8h;
typedef __attribute__((ext_vector_type(8)))  float    v8f;
typedef __attribute__((ext_vector_type(4)))  float    f32x4;

#define BATCH 64
#define MDIM  512
#define NDIM  512
#define KDIM  256
#define BETA_INV (1.0f / 1000.0f)

#define LDS_ROWH 136   // 128 payload halves + 8 pad halves = 272B row stride

// wave32 all-lanes sum
__device__ __forceinline__ float wave_sum(float x) {
#pragma unroll
    for (int off = 16; off > 0; off >>= 1) x += __shfl_xor(x, off, 32);
    return x;
}

// Async copy 16B global -> LDS (per-lane), tracked by ASYNCcnt.
__device__ __forceinline__ void async_ld16(unsigned lds_addr, const _Float16* g) {
    asm volatile("global_load_async_to_lds_b128 %0, %1, off"
                 :: "v"(lds_addr), "v"((unsigned long long)(size_t)g)
                 : "memory");
}

// --- kernel 0: zero the 4*BATCH accumulators (256 floats) ------------------
__global__ void pot_init_kernel(float* __restrict__ acc) {
    acc[threadIdx.x] = 0.0f;
}

// --- kernel 1: fused f32->f16 convert + squared row norms ------------------
// One wave per row; lane l handles contiguous elements [8l, 8l+8).
__global__ __launch_bounds__(256) void pot_cvt_norm_kernel(
    const float* __restrict__ v, const float* __restrict__ a,
    _Float16* __restrict__ v16, _Float16* __restrict__ a16,
    float* __restrict__ vnorm, float* __restrict__ anorm) {
    int wave  = threadIdx.x >> 5;
    int lane  = threadIdx.x & 31;
    int rid   = blockIdx.x * 8 + wave;      // 0 .. 2*BATCH*512 - 1
    int which = rid >> 15;                  // 0 = v, 1 = a
    int row   = rid & 32767;                // b*512 + i
    const float* src = (which ? a : v) + (size_t)row * KDIM + lane * 8;
    const f32x4* sp = (const f32x4*)src;
    f32x4 f0 = sp[0], f1 = sp[1];
    v8h h;
    h[0] = (_Float16)f0[0]; h[1] = (_Float16)f0[1];
    h[2] = (_Float16)f0[2]; h[3] = (_Float16)f0[3];
    h[4] = (_Float16)f1[0]; h[5] = (_Float16)f1[1];
    h[6] = (_Float16)f1[2]; h[7] = (_Float16)f1[3];
    _Float16* dst = (which ? a16 : v16) + (size_t)row * KDIM + lane * 8;
    *(v8h*)dst = h;
    float s = f0[0]*f0[0] + f0[1]*f0[1] + f0[2]*f0[2] + f0[3]*f0[3]
            + f1[0]*f1[0] + f1[1]*f1[1] + f1[2]*f1[2] + f1[3]*f1[3];
    s = wave_sum(s);
    if (lane == 0) (which ? anorm : vnorm)[row] = s;
}

// Epilogue for one 16x16 accumulator: C/D layout -> VGPR r = (M=r+8*hi, N=lr)
__device__ __forceinline__ void pot_epilogue(
    const v8f& acc, const float* nvp, float na,
    bool rowflag, bool colflag, int hi, int lr,
    float& sS, float& sCT, float& sR, float& sC) {
#pragma unroll
    for (int r = 0; r < 8; ++r) {
        float Cv = 0.5f * (nvp[r] + na) - acc[r];
        float tv = __expf(-Cv * BETA_INV);
        sS  += tv;
        sCT += Cv * tv;
        if (rowflag && hi == 0 && r == 0) sR += tv;  // global row 0
        if (colflag && lr == 0)           sC += tv;  // global col 0
    }
}

// Build a 16-half A/B fragment from two 16B LDS chunks.
__device__ __forceinline__ v16h frag16(const _Float16* p0, const _Float16* p1) {
    v8h lo = *(const v8h*)p0;
    v8h hi = *(const v8h*)p1;
    return __builtin_shufflevector(lo, hi,
        0,1,2,3,4,5,6,7,8,9,10,11,12,13,14,15);
}

// --- kernel 2: WMMA tile kernel --------------------------------------------
// Block = 8 waves = 64x128 output region; wave = 32x32 tile.
// Grid: (512/64) x (512/128) x 64 = 2048 blocks.
__global__ __launch_bounds__(256) void pot_tile_kernel(
    const _Float16* __restrict__ v16, const _Float16* __restrict__ a16,
    const float* __restrict__ vnorm, const float* __restrict__ anorm,
    float* __restrict__ accS, float* __restrict__ accCT,
    float* __restrict__ accRS, float* __restrict__ accCS) {
    __shared__ _Float16 shA[64 * LDS_ROWH];    // 17408 B
    __shared__ _Float16 shB[128 * LDS_ROWH];   // 34816 B
    int tid  = threadIdx.x;
    int wave = tid >> 5;
    int lane = tid & 31;
    int bid  = blockIdx.x;
    int ib = bid & 7;               // M block (64 rows)
    int jb = (bid >> 3) & 3;        // N block (128 rows)
    int b  = bid >> 5;              // batch
    int i0b = ib * 64, j0b = jb * 128;
    int wi = wave & 1;              // wave's M sub-tile (32 rows)
    int wj = wave >> 1;             // wave's N sub-tile (32 rows)
    int lr = lane & 15;
    int hi = lane >> 4;

    // LDS byte addresses for the async engine (ISA: LDS addr = flat[31:0]).
    unsigned ldsA = (unsigned)(size_t)&shA[0];
    unsigned ldsB = (unsigned)(size_t)&shB[0];

    const _Float16* vbase = v16 + (size_t)(b * MDIM + i0b) * KDIM;
    const _Float16* abase = a16 + (size_t)(b * NDIM + j0b) * KDIM;

    v8f acc00 = {}, acc01 = {}, acc10 = {}, acc11 = {};

#pragma unroll
    for (int kc = 0; kc < 2; ++kc) {          // two K=128 chunks
        if (kc) __syncthreads();              // protect LDS being re-staged
        int kbase = kc * 128;
        // ---- stage A panel: 64 rows x 128 halves = 1024 x 16B chunks ----
#pragma unroll
        for (int r = 0; r < 4; ++r) {
            int c = r * 256 + tid;            // 16B-chunk id
            int row = c >> 4, col = c & 15;   // 16 chunks per row
            async_ld16(ldsA + (unsigned)(row * (LDS_ROWH * 2) + col * 16),
                       vbase + (size_t)row * KDIM + kbase + col * 8);
        }
        // ---- stage B panel: 128 rows x 128 halves = 2048 x 16B chunks ----
#pragma unroll
        for (int r = 0; r < 8; ++r) {
            int c = r * 256 + tid;
            int row = c >> 4, col = c & 15;
            async_ld16(ldsB + (unsigned)(row * (LDS_ROWH * 2) + col * 16),
                       abase + (size_t)row * KDIM + kbase + col * 8);
        }
        asm volatile("s_wait_asynccnt 0x0" ::: "memory");
        __syncthreads();

        // ---- compute: 4 k-steps of K=32 from LDS ----
        const _Float16* vr0 = shA + (wi * 32 + lr) * LDS_ROWH;
        const _Float16* vr1 = vr0 + 16 * LDS_ROWH;
        const _Float16* ar0 = shB + (wj * 32 + lr) * LDS_ROWH;
        const _Float16* ar1 = ar0 + 16 * LDS_ROWH;
#pragma unroll
        for (int ks = 0; ks < 4; ++ks) {
            int k0 = ks * 32;
            // A frags: lane<16 -> K {k0..+7, k0+16..+23}; lanes>=16 +8
            v16h af0 = frag16(vr0 + k0 + 8 * hi, vr0 + k0 + 16 + 8 * hi);
            v16h af1 = frag16(vr1 + k0 + 8 * hi, vr1 + k0 + 16 + 8 * hi);
            // B frags: column N=lr, same K order as A frags
            v16h bf0 = frag16(ar0 + k0 + 8 * hi, ar0 + k0 + 16 + 8 * hi);
            v16h bf1 = frag16(ar1 + k0 + 8 * hi, ar1 + k0 + 16 + 8 * hi);
            acc00 = __builtin_amdgcn_wmma_f32_16x16x32_f16(
                false, af0, false, bf0, (short)0, acc00, false, false);
            acc01 = __builtin_amdgcn_wmma_f32_16x16x32_f16(
                false, af0, false, bf1, (short)0, acc01, false, false);
            acc10 = __builtin_amdgcn_wmma_f32_16x16x32_f16(
                false, af1, false, bf0, (short)0, acc10, false, false);
            acc11 = __builtin_amdgcn_wmma_f32_16x16x32_f16(
                false, af1, false, bf1, (short)0, acc11, false, false);
        }
    }

    int i0 = i0b + wi * 32;
    int j0 = j0b + wj * 32;
    float na0 = anorm[(size_t)b * NDIM + j0 + lr];
    float na1 = anorm[(size_t)b * NDIM + j0 + 16 + lr];
    const float* nv0 = vnorm + (size_t)b * MDIM + i0 + 8 * hi;
    const float* nv1 = nv0 + 16;

    float sS = 0.0f, sCT = 0.0f, sR = 0.0f, sC = 0.0f;
    pot_epilogue(acc00, nv0, na0, true,  true,  hi, lr, sS, sCT, sR, sC);
    pot_epilogue(acc01, nv0, na1, true,  false, hi, lr, sS, sCT, sR, sC);
    pot_epilogue(acc10, nv1, na0, false, true,  hi, lr, sS, sCT, sR, sC);
    pot_epilogue(acc11, nv1, na1, false, false, hi, lr, sS, sCT, sR, sC);

    sS  = wave_sum(sS);
    sCT = wave_sum(sCT);
    sR  = wave_sum(sR);
    sC  = wave_sum(sC);
    if (lane == 0) {
        atomicAdd(&accS[b],  sS);
        atomicAdd(&accCT[b], sCT);
        if (i0 == 0) atomicAdd(&accRS[b], sR);
        if (j0 == 0) atomicAdd(&accCS[b], sC);
    }
}

// --- kernel 3: scalar recurrence + mean over batch -------------------------
__global__ void pot_final_kernel(const float* __restrict__ accS,
                                 const float* __restrict__ accCT,
                                 const float* __restrict__ accRS,
                                 const float* __restrict__ accCS,
                                 float* __restrict__ out) {
    __shared__ float red[2];
    int tid = threadIdx.x;                  // 64 threads = batch index
    float S  = accS[tid];
    float CT = accCT[tid];
    float rs = accRS[tid];
    float cs = accCS[tid];
    const float a0 = 1.0f / 512.0f, b0 = 1.0f / 512.0f, s = 512.0f;
    float c = s / S;
#pragma unroll
    for (int it = 0; it < 10; ++it) {
        float ka = fminf(a0 / (c * rs), 1.0f);
        float kb = fminf(b0 / (ka * c * cs), 1.0f);
        c = s * ka * kb / S;
    }
    float Dv = c * CT;
    Dv = wave_sum(Dv);                      // per-wave32 partial sums
    if ((tid & 31) == 0) red[tid >> 5] = Dv;
    __syncthreads();
    if (tid == 0) out[0] = (red[0] + red[1]) * (1.0f / 64.0f);
}

// ---------------------------------------------------------------------------
extern "C" void kernel_launch(void* const* d_in, const int* in_sizes, int n_in,
                              void* d_out, int out_size, void* d_ws, size_t ws_size,
                              hipStream_t stream) {
    const float* v = (const float*)d_in[0];   // [64,512,256] f32
    const float* a = (const float*)d_in[1];   // [64,512,256] f32

    // workspace: v16[8M halves] | a16[8M halves] | vnorm[32K] | anorm[32K]
    //            | accS/CT/RS/CS[64 each]   (~33 MB total)
    _Float16* v16  = (_Float16*)d_ws;
    _Float16* a16  = v16 + (size_t)BATCH * MDIM * KDIM;
    float* vnorm   = (float*)(a16 + (size_t)BATCH * NDIM * KDIM);
    float* anorm   = vnorm + BATCH * MDIM;
    float* accS    = anorm + BATCH * NDIM;
    float* accCT   = accS + BATCH;
    float* accRS   = accCT + BATCH;
    float* accCS   = accRS + BATCH;

    pot_init_kernel<<<1, 256, 0, stream>>>(accS);   // zeros all 4 acc arrays
    pot_cvt_norm_kernel<<<8192, 256, 0, stream>>>(v, a, v16, a16, vnorm, anorm);
    pot_tile_kernel<<<2048, 256, 0, stream>>>(v16, a16, vnorm, anorm,
                                              accS, accCT, accRS, accCS);
    pot_final_kernel<<<1, 64, 0, stream>>>(accS, accCT, accRS, accCS,
                                           (float*)d_out);
}
